// CompPCFG_1657857376435
// MI455X (gfx1250) — compile-verified, matrix operand order checked by reading, not run on hardware
//
#include <hip/hip_runtime.h>
#include <math.h>

#define B_ 16
#define N_ 25
#define V_ 10000
#define WDIM 512
#define HDIM 512
#define ZDIM 64
#define SDIM 256
#define T_ 60
#define NT_ 30
#define S_ 90
#define NEGC (-1.0e9f)

typedef __attribute__((ext_vector_type(16))) __bf16 v16bf;
typedef __attribute__((ext_vector_type(8)))  float v8f;

__device__ __forceinline__ unsigned short f2bf_bits(float f) {
  union { float f; unsigned u; } v; v.f = f;
  return (unsigned short)((v.u + 0x7FFFu + ((v.u >> 16) & 1u)) >> 16);
}
__device__ __forceinline__ __bf16 f2bf(float f) {
  union { unsigned short s; __bf16 b; } o; o.s = f2bf_bits(f); return o.b;
}

// Branch-free fast sigmoid/tanh on v_exp_f32 + v_rcp_f32 (no precise-div sequences).
__device__ __forceinline__ float fsigmoid(float x) {
  return __builtin_amdgcn_rcpf(1.0f + __expf(-x));
}
__device__ __forceinline__ float ftanh(float x) {
  return 2.0f * __builtin_amdgcn_rcpf(1.0f + __expf(-2.0f * x)) - 1.0f;
}

// K offset of element e within a 32-K-slab for a 16-bit WMMA fragment.
// ISA 16x32 layout: per lane-half, K is contiguous: e<8 -> half*8+e ; e>=8 -> 16+half*8+(e-8)
__device__ __forceinline__ int frag_k(int e, int half) {
  return (e < 8) ? (half * 8 + e) : (16 + half * 8 + (e - 8));
}

// A fragment: 4 x float4 contiguous loads, convert to bf16. (32-bit indexing)
__device__ __forceinline__ v16bf load_a_frag(const float* __restrict__ A, int lda,
                                             int row, int k0, int half) {
  const float* p = A + row * lda + k0 + half * 8;
  float4 x0 = ((const float4*)p)[0];
  float4 x1 = ((const float4*)p)[1];
  float4 y0 = ((const float4*)(p + 16))[0];
  float4 y1 = ((const float4*)(p + 16))[1];
  v16bf a;
  a[0] = f2bf(x0.x); a[1] = f2bf(x0.y); a[2] = f2bf(x0.z); a[3] = f2bf(x0.w);
  a[4] = f2bf(x1.x); a[5] = f2bf(x1.y); a[6] = f2bf(x1.z); a[7] = f2bf(x1.w);
  a[8] = f2bf(y0.x); a[9] = f2bf(y0.y); a[10] = f2bf(y0.z); a[11] = f2bf(y0.w);
  a[12] = f2bf(y1.x); a[13] = f2bf(y1.y); a[14] = f2bf(y1.z); a[15] = f2bf(y1.w);
  return a;
}

// Swizzled-B fragment: one contiguous 32B vector load. (32-bit indexing)
__device__ __forceinline__ v16bf load_b_swz(const unsigned short* __restrict__ Bsw,
                                            int kb, int Npad, int col, int half) {
  int idx = (((kb * Npad + col) << 1) + half) << 4;
  return *(const v16bf*)(Bsw + idx);
}

// ---- pre-swizzle a (K,N) f32 weight into WMMA-native bf16 fragments --------
__global__ void k_swz_b(const float* __restrict__ Bm, unsigned short* __restrict__ out,
                        int K, int N, int Npad) {
  long i = (long)blockIdx.x * 256 + threadIdx.x;
  long total = (long)(K >> 5) * Npad * 32;
  if (i >= total) return;
  int e = (int)(i & 15);
  long rest = i >> 4;
  int half = (int)(rest & 1); rest >>= 1;
  int col = (int)(rest % Npad);
  int kb  = (int)(rest / Npad);
  int k = (kb << 5) + frag_k(e, half);
  float v = (col < N) ? Bm[(size_t)k * N + col] : 0.0f;
  out[i] = f2bf_bits(v);
}

// ------- generic WMMA GEMM: C = act(A@Bsw + bias) [+ res]  (compile-time variant)
template <int ACT, bool RES>
__global__ void k_gemm_bf16(const float* __restrict__ A, const unsigned short* __restrict__ Bsw,
                            const float* __restrict__ bias, const float* __restrict__ res,
                            float* __restrict__ C, int M, int Nc, int Npad, int K) {
  int lane = threadIdx.x & 31, wave = threadIdx.x >> 5;
  int half = lane >> 4, l16 = lane & 15;
  int tn = blockIdx.x * 4 + wave;
  int row0 = blockIdx.y * 16;
  if (tn * 16 >= Npad) return;
  int col = tn * 16 + l16;
  int colc = (col < Nc) ? col : 0;       // clamped index (no predicated loads)
  v8f acc;
  float bv = bias[colc];
#pragma unroll
  for (int r = 0; r < 8; ++r) acc[r] = bv;
  int arow = row0 + l16;
  for (int k0 = 0; k0 < K; k0 += 32) {
    v16bf af = load_a_frag(A, K, arow, k0, half);
    v16bf bf = load_b_swz(Bsw, k0 >> 5, Npad, col, half);
    acc = __builtin_amdgcn_wmma_f32_16x16x32_bf16(false, af, false, bf,
                                                  (short)0, acc, false, false);
  }
  bool ok = (col < Nc);
#pragma unroll
  for (int r = 0; r < 8; ++r) {
    int m = row0 + r + 8 * half;
    float v = acc[r];
    if (ACT) v = fmaxf(v, 0.0f);
    if (RES) v += res[m * Nc + colc];
    if (ok) C[m * Nc + col] = v;
  }
}

// ---------------- LSTM recurrent step (WMMA, 4 gate tiles per wave) ----------
__global__ void k_lstm_step(const float* __restrict__ xi,      // (16,2048) this t
                            const float* __restrict__ hprev,   // (16,512)
                            const float* __restrict__ cprev,   // (16,512)
                            const unsigned short* __restrict__ whh, // swizzled bf16 (512,2048)
                            float* __restrict__ hout, float* __restrict__ cout) {
  int lane = threadIdx.x & 31, wave = threadIdx.x >> 5;
  int half = lane >> 4, l16 = lane & 15;
  int cg = blockIdx.x * 4 + wave;     // 0..31 column group within H
  int col = cg * 16 + l16;
  v8f acc[4];
#pragma unroll
  for (int g = 0; g < 4; ++g)
#pragma unroll
    for (int r = 0; r < 8; ++r) acc[g][r] = xi[(r + 8 * half) * 2048 + g * 512 + col];
  for (int k0 = 0; k0 < 512; k0 += 32) {
    v16bf af = load_a_frag(hprev, 512, l16, k0, half);
#pragma unroll
    for (int g = 0; g < 4; ++g) {
      v16bf bf = load_b_swz(whh, k0 >> 5, 2048, g * 512 + col, half);
      acc[g] = __builtin_amdgcn_wmma_f32_16x16x32_bf16(false, af, false, bf,
                                                       (short)0, acc[g], false, false);
    }
  }
#pragma unroll
  for (int r = 0; r < 8; ++r) {
    int m = r + 8 * half;
    float iv = fsigmoid(acc[0][r]);
    float fv = fsigmoid(acc[1][r]);
    float gv = ftanh(acc[2][r]);
    float ov = fsigmoid(acc[3][r]);
    float cn = fv * cprev[m * 512 + col] + iv * gv;
    cout[m * 512 + col] = cn;
    hout[m * 512 + col] = ov * ftanh(cn);
  }
}

// ---------------- vocab logits GEMM fused with streaming logsumexp ----------
__global__ void k_vocab_lse(const float* __restrict__ H,            // (960,256)
                            const unsigned short* __restrict__ Wsw, // swizzled (256,10000)
                            const float* __restrict__ bias,         // (10000)
                            float* __restrict__ lse) {              // (960)
  __shared__ float RM[1024], RS[1024];
  int lane = threadIdx.x & 31, wave = threadIdx.x >> 5;
  int half = lane >> 4, l16 = lane & 15;
  int row0 = blockIdx.x * 16;
  v16bf afr[8];
#pragma unroll
  for (int kk = 0; kk < 8; ++kk) afr[kk] = load_a_frag(H, 256, row0 + l16, kk * 32, half);
  float m[8], s[8];
#pragma unroll
  for (int r = 0; r < 8; ++r) { m[r] = -3.0e38f; s[r] = 0.0f; }
  for (int tIdx = wave; tIdx < 625; tIdx += 4) {   // 625*16 == 10000
    int col = tIdx * 16 + l16;
    v8f acc;
    float bv = bias[col];
#pragma unroll
    for (int r = 0; r < 8; ++r) acc[r] = bv;
#pragma unroll
    for (int kk = 0; kk < 8; ++kk) {
      v16bf bf = load_b_swz(Wsw, kk, V_, col, half);
      acc = __builtin_amdgcn_wmma_f32_16x16x32_bf16(false, afr[kk], false, bf,
                                                    (short)0, acc, false, false);
    }
#pragma unroll
    for (int r = 0; r < 8; ++r) {
      float v = acc[r];
      if (v > m[r]) { s[r] = s[r] * __expf(m[r] - v) + 1.0f; m[r] = v; }
      else s[r] += __expf(v - m[r]);
    }
  }
#pragma unroll
  for (int r = 0; r < 8; ++r) {
    int idx = ((wave * 2 + half) * 8 + r) * 16 + l16;
    RM[idx] = m[r]; RS[idx] = s[r];
  }
  __syncthreads();
  if (threadIdx.x < 16) {
    int rloc = threadIdx.x, h2 = rloc >> 3, rr = rloc & 7;
    float M = -3.0e38f, S = 0.0f;
    for (int w = 0; w < 4; ++w)
      for (int l = 0; l < 16; ++l) {
        int idx = ((w * 2 + h2) * 8 + rr) * 16 + l;
        float mm = RM[idx], ss = RS[idx];
        if (mm > M) { S = S * __expf(M - mm) + ss; M = mm; }
        else S += ss * __expf(mm - M);
      }
    lse[row0 + rloc] = M + __logf(S);
  }
}

// ---------------- small helper kernels --------------------------------------
__global__ void k_embed(const int* __restrict__ x, const float* __restrict__ emb,
                        float* __restrict__ out) {
  int row = blockIdx.x;                 // n*B + b
  int n = row / B_, b = row % B_;
  int idx = x[b * N_ + n];
  for (int w = threadIdx.x; w < WDIM; w += blockDim.x)
    out[row * WDIM + w] = emb[(size_t)idx * WDIM + w];
}

__global__ void k_zero(float* p, int n) {
  int i = blockIdx.x * 256 + threadIdx.x;
  if (i < n) p[i] = 0.0f;
}

__global__ void k_pool_enc(const float* __restrict__ hsf, const float* __restrict__ hsb,
                           const float* __restrict__ eps, const float* __restrict__ encw,
                           const float* __restrict__ encb, float* __restrict__ z,
                           float* __restrict__ dout) {
  __shared__ float hcat[16 * 1024];
  __shared__ float P[16 * 128];
  __shared__ float KLM[16 * 64];
  for (int idx = threadIdx.x; idx < 16 * 1024; idx += blockDim.x) {
    int b = idx >> 10, j = idx & 1023;
    float mx = -3.0e38f;
    if (j < 512) { for (int t = 0; t < N_; ++t) mx = fmaxf(mx, hsf[(t * 16 + b) * 512 + j]); }
    else { int jj = j - 512; for (int t = 0; t < N_; ++t) mx = fmaxf(mx, hsb[(t * 16 + b) * 512 + jj]); }
    hcat[idx] = mx;
  }
  __syncthreads();
  if (threadIdx.x < 128) {
    int p = threadIdx.x;
    for (int b = 0; b < 16; ++b) {
      float s = encb[p];
      for (int j = 0; j < 1024; ++j) s += hcat[b * 1024 + j] * encw[j * 128 + p];
      P[b * 128 + p] = s;
    }
  }
  __syncthreads();
  if (threadIdx.x < 64) {
    int zd = threadIdx.x;
    for (int b = 0; b < 16; ++b) {
      float mean = P[b * 128 + zd], lv = P[b * 128 + 64 + zd];
      z[b * 64 + zd] = __expf(0.5f * lv) * eps[b * 64 + zd] + mean;
      KLM[b * 64 + zd] = -0.5f * (lv - mean * mean - __expf(lv) + 1.0f);
    }
  }
  __syncthreads();
  if (threadIdx.x < 16) {
    int b = threadIdx.x; float s = 0.0f;
    for (int zd = 0; zd < 64; ++zd) s += KLM[b * 64 + zd];
    dout[16 + b] = s;
  }
}

__global__ void k_concat(const float* __restrict__ emb, int perB, const float* __restrict__ z,
                         float* __restrict__ out, int rows) {
  int row = blockIdx.x;
  if (row >= rows) return;
  int b = row / perB, e = row % perB;
  for (int j = threadIdx.x; j < SDIM + ZDIM; j += blockDim.x)
    out[row * 320 + j] = (j < SDIM) ? emb[e * SDIM + j] : z[b * ZDIM + (j - SDIM)];
}

__global__ void k_root_mlp(const float* __restrict__ rin, const float* __restrict__ w1,
                           const float* __restrict__ b1, const float* __restrict__ rw,
                           const float* __restrict__ rb, const float* __restrict__ w2,
                           const float* __restrict__ b2, float* __restrict__ out) {
  __shared__ float X[16 * 320];
  __shared__ float Hh[16 * 256];
  __shared__ float Aa[16 * 256];
  __shared__ float H2[16 * 256];
  __shared__ float O[16 * 30];
  int j = threadIdx.x;
  for (int i = j; i < 16 * 320; i += blockDim.x) X[i] = rin[i];
  __syncthreads();
  for (int m = 0; m < 16; ++m) {
    float s = b1[j];
    for (int k = 0; k < 320; ++k) s += X[m * 320 + k] * w1[k * 256 + j];
    Hh[m * 256 + j] = s;
  }
  __syncthreads();
  for (int L = 0; L < 2; ++L) {
    const float* w0 = rw + (2 * L) * 256 * 256;      const float* bb0 = rb + (2 * L) * 256;
    const float* w1r = rw + (2 * L + 1) * 256 * 256; const float* bb1 = rb + (2 * L + 1) * 256;
    for (int m = 0; m < 16; ++m) {
      float s = bb0[j];
      for (int k = 0; k < 256; ++k) s += Hh[m * 256 + k] * w0[k * 256 + j];
      Aa[m * 256 + j] = fmaxf(s, 0.0f);
    }
    __syncthreads();
    for (int m = 0; m < 16; ++m) {
      float s = bb1[j];
      for (int k = 0; k < 256; ++k) s += Aa[m * 256 + k] * w1r[k * 256 + j];
      H2[m * 256 + j] = fmaxf(s, 0.0f) + Hh[m * 256 + j];
    }
    __syncthreads();
    for (int m = 0; m < 16; ++m) Hh[m * 256 + j] = H2[m * 256 + j];
    __syncthreads();
  }
  if (j < 30)
    for (int m = 0; m < 16; ++m) {
      float s = b2[j];
      for (int k = 0; k < 256; ++k) s += Hh[m * 256 + k] * w2[k * 30 + j];
      O[m * 30 + j] = s;
    }
  __syncthreads();
  if (j < 16) {
    float mx = -3.0e38f;
    for (int A = 0; A < 30; ++A) mx = fmaxf(mx, O[j * 30 + A]);
    float s = 0.0f;
    for (int A = 0; A < 30; ++A) s += __expf(O[j * 30 + A] - mx);
    float l = mx + __logf(s);
    for (int A = 0; A < 30; ++A) out[j * 30 + A] = O[j * 30 + A] - l;
  }
}

__global__ void k_unary(const float* __restrict__ H, const float* __restrict__ W2,
                        const float* __restrict__ b2, const float* __restrict__ lse,
                        const int* __restrict__ x, float* __restrict__ unary) {
  int bn = blockIdx.x;
  int b = bn / N_, n = bn % N_;
  int xv = x[b * N_ + n];
  for (int t = threadIdx.x; t < T_; t += blockDim.x) {
    const float* h = H + (b * T_ + t) * 256;
    float s = b2[xv];
    for (int k = 0; k < 256; ++k) s += h[k] * W2[k * V_ + xv];
    unary[(b * N_ + n) * T_ + t] = s - lse[b * T_ + t];
  }
}

__global__ void k_lsm_rows(float* __restrict__ data, int cols) {
  __shared__ float red[256];
  float* row = data + blockIdx.x * cols;
  float m = -3.0e38f;
  for (int c = threadIdx.x; c < cols; c += 256) m = fmaxf(m, row[c]);
  red[threadIdx.x] = m; __syncthreads();
  for (int o = 128; o > 0; o >>= 1) {
    if (threadIdx.x < o) red[threadIdx.x] = fmaxf(red[threadIdx.x], red[threadIdx.x + o]);
    __syncthreads();
  }
  m = red[0]; __syncthreads();
  float s = 0.0f;
  for (int c = threadIdx.x; c < cols; c += 256) s += __expf(row[c] - m);
  red[threadIdx.x] = s; __syncthreads();
  for (int o = 128; o > 0; o >>= 1) {
    if (threadIdx.x < o) red[threadIdx.x] += red[threadIdx.x + o];
    __syncthreads();
  }
  float l = m + __logf(red[0]);
  for (int c = threadIdx.x; c < cols; c += 256) row[c] -= l;
}

__global__ void k_chart_init(float* __restrict__ chart, const float* __restrict__ unary) {
  int i = blockIdx.x * 256 + threadIdx.x;
  if (i >= B_ * N_ * N_ * S_) return;
  int sidx = i % S_; int rest = i / S_;
  int jc = rest % N_; rest /= N_;
  int ir = rest % N_; int b = rest / N_;
  float v = NEGC;
  if (ir == jc && sidx >= NT_) v = unary[(b * N_ + ir) * T_ + (sidx - NT_)];
  chart[i] = v;
}

__global__ void k_inside(float* __restrict__ chart, const float* __restrict__ rule, int w) {
  __shared__ float Lh[24 * 90];
  __shared__ float Rh[24 * 90];
  __shared__ float M2[8100];
  __shared__ float RM[256], RS[256];
  int i = blockIdx.x, b = blockIdx.y;
  int ns = w - 1;
  for (int idx = threadIdx.x; idx < ns * S_; idx += blockDim.x) {
    int u = idx / S_, s = idx % S_;
    Lh[idx] = chart[(((b * N_ + i) * N_) + (i + u)) * S_ + s];
    Rh[idx] = chart[(((b * N_ + (i + u + 1)) * N_) + (i + w - 1)) * S_ + s];
  }
  __syncthreads();
  for (int p = threadIdx.x; p < S_ * S_; p += blockDim.x) {
    int s1 = p / S_, s2 = p % S_;
    float mx = -3.0e38f;
    for (int u = 0; u < ns; ++u) mx = fmaxf(mx, Lh[u * S_ + s1] + Rh[u * S_ + s2]);
    float ss = 0.0f;
    for (int u = 0; u < ns; ++u) ss += __expf(Lh[u * S_ + s1] + Rh[u * S_ + s2] - mx);
    M2[p] = mx + __logf(ss);
  }
  __syncthreads();
  for (int A = 0; A < NT_; ++A) {
    const float* rr = rule + (b * NT_ + A) * (S_ * S_);
    float mx = -3.0e38f, ssum = 0.0f;
    for (int p = threadIdx.x; p < S_ * S_; p += blockDim.x) {
      float v = rr[p] + M2[p];
      if (v > mx) { ssum = ssum * __expf(mx - v) + 1.0f; mx = v; }
      else ssum += __expf(v - mx);
    }
    RM[threadIdx.x] = mx; RS[threadIdx.x] = ssum;
    __syncthreads();
    if (threadIdx.x == 0) {
      float M = -3.0e38f, S = 0.0f;
      for (int t = 0; t < 256; ++t) {
        float mm = RM[t], ss2 = RS[t];
        if (mm > M) { S = S * __expf(M - mm) + ss2; M = mm; }
        else S += ss2 * __expf(mm - M);
      }
      chart[(((b * N_ + i) * N_) + (i + w - 1)) * S_ + A] = M + __logf(S);
    }
    __syncthreads();
  }
}

__global__ void k_final(const float* __restrict__ chart, const float* __restrict__ roots,
                        float* __restrict__ dout) {
  __shared__ float buf[32];
  int b = blockIdx.x, A = threadIdx.x;
  buf[A] = (A < NT_) ? roots[b * NT_ + A] + chart[(((b * N_) * N_) + (N_ - 1)) * S_ + A]
                     : -3.0e38f;
  __syncthreads();
  if (A == 0) {
    float mx = -3.0e38f;
    for (int t = 0; t < NT_; ++t) mx = fmaxf(mx, buf[t]);
    float s = 0.0f;
    for (int t = 0; t < NT_; ++t) s += __expf(buf[t] - mx);
    dout[b] = -(mx + __logf(s));
  }
}

// ---------------------------------------------------------------------------
static inline int swz_blocks(int K, int N, int Npad) {
  long total = (long)(K >> 5) * Npad * 32;
  return (int)((total + 255) / 256);
}

extern "C" void kernel_launch(void* const* d_in, const int* in_sizes, int n_in,
                              void* d_out, int out_size, void* d_ws, size_t ws_size,
                              hipStream_t stream) {
  const int*   x        = (const int*)  d_in[0];
  const float* eps      = (const float*)d_in[1];
  const float* enc_emb  = (const float*)d_in[2];
  const float* f_wih    = (const float*)d_in[3];
  const float* f_whh    = (const float*)d_in[4];
  const float* f_b      = (const float*)d_in[5];
  const float* b_wih    = (const float*)d_in[6];
  const float* b_whh    = (const float*)d_in[7];
  const float* b_b      = (const float*)d_in[8];
  const float* encp_w   = (const float*)d_in[9];
  const float* encp_b   = (const float*)d_in[10];
  const float* t_emb    = (const float*)d_in[11];
  const float* nt_emb   = (const float*)d_in[12];
  const float* root_emb = (const float*)d_in[13];
  const float* rule_w   = (const float*)d_in[14];
  const float* rule_b   = (const float*)d_in[15];
  const float* root_w1  = (const float*)d_in[16];
  const float* root_b1  = (const float*)d_in[17];
  const float* root_rw  = (const float*)d_in[18];
  const float* root_rb  = (const float*)d_in[19];
  const float* root_w2  = (const float*)d_in[20];
  const float* root_b2  = (const float*)d_in[21];
  const float* voc_w1   = (const float*)d_in[22];
  const float* voc_b1   = (const float*)d_in[23];
  const float* voc_rw   = (const float*)d_in[24];
  const float* voc_rb   = (const float*)d_in[25];
  const float* voc_w2   = (const float*)d_in[26];
  const float* voc_b2   = (const float*)d_in[27];

  char* ws = (char*)d_ws;
  size_t off = 0;
  auto alloc = [&](size_t bytes) -> char* {
    char* p = ws + off;
    off = (off + bytes + 255) & ~(size_t)255;
    return p;
  };
  float*          emb    = (float*)alloc(400 * 512 * 4);
  unsigned short* whhFs  = (unsigned short*)alloc((size_t)512 * 2048 * 2);
  unsigned short* whhBs  = (unsigned short*)alloc((size_t)512 * 2048 * 2);
  unsigned short* wihFs  = (unsigned short*)alloc((size_t)512 * 2048 * 2);
  unsigned short* wihBs  = (unsigned short*)alloc((size_t)512 * 2048 * 2);
  unsigned short* vw1s   = (unsigned short*)alloc((size_t)320 * 256 * 2);
  unsigned short* vrws   = (unsigned short*)alloc((size_t)4 * 256 * 256 * 2);
  unsigned short* vw2s   = (unsigned short*)alloc((size_t)256 * 10000 * 2);
  unsigned short* rulsw  = (unsigned short*)alloc((size_t)320 * 8112 * 2);  // Npad=8112
  float*          xiF    = (float*)alloc((size_t)400 * 2048 * 4);
  float*          xiB    = (float*)alloc((size_t)400 * 2048 * 4);
  float*          hsF    = (float*)alloc(25 * 16 * 512 * 4);
  float*          hsB    = (float*)alloc(25 * 16 * 512 * 4);
  float*          h0     = (float*)alloc(16 * 512 * 4);
  float*          cF0    = (float*)alloc(16 * 512 * 4);
  float*          cF1    = (float*)alloc(16 * 512 * 4);
  float*          cB0    = (float*)alloc(16 * 512 * 4);
  float*          cB1    = (float*)alloc(16 * 512 * 4);
  float*          z      = (float*)alloc(16 * 64 * 4);
  float*          tin    = (float*)alloc((size_t)960 * 320 * 4);
  float*          ntin   = (float*)alloc((size_t)480 * 320 * 4);
  float*          rin    = (float*)alloc(16 * 320 * 4);
  float*          hA     = (float*)alloc((size_t)960 * 256 * 4);
  float*          aA     = (float*)alloc((size_t)960 * 256 * 4);
  float*          hB2    = (float*)alloc((size_t)960 * 256 * 4);
  float*          lse    = (float*)alloc(960 * 4);
  float*          un     = (float*)alloc(16 * 25 * 60 * 4);
  float*          rule   = (float*)alloc((size_t)480 * 8100 * 4);
  float*          roots  = (float*)alloc(16 * 30 * 4);
  float*          chart  = (float*)alloc((size_t)16 * 25 * 25 * 90 * 4);
  (void)ws_size; (void)in_sizes; (void)n_in; (void)out_size;

  // 0) pre-swizzle all WMMA B-operands into fragment-native bf16 (one-time)
  k_swz_b<<<swz_blocks(512, 2048, 2048), 256, 0, stream>>>(f_whh, whhFs, 512, 2048, 2048);
  k_swz_b<<<swz_blocks(512, 2048, 2048), 256, 0, stream>>>(b_whh, whhBs, 512, 2048, 2048);
  k_swz_b<<<swz_blocks(512, 2048, 2048), 256, 0, stream>>>(f_wih, wihFs, 512, 2048, 2048);
  k_swz_b<<<swz_blocks(512, 2048, 2048), 256, 0, stream>>>(b_wih, wihBs, 512, 2048, 2048);
  k_swz_b<<<swz_blocks(320, 256, 256), 256, 0, stream>>>(voc_w1, vw1s, 320, 256, 256);
  for (int i = 0; i < 4; ++i)
    k_swz_b<<<swz_blocks(256, 256, 256), 256, 0, stream>>>(voc_rw + (size_t)i * 65536,
                                                           vrws + (size_t)i * 65536, 256, 256, 256);
  k_swz_b<<<swz_blocks(256, 10000, 10000), 256, 0, stream>>>(voc_w2, vw2s, 256, 10000, 10000);
  k_swz_b<<<swz_blocks(320, 8100, 8112), 256, 0, stream>>>(rule_w, rulsw, 320, 8100, 8112);

  // 1) embedding gather (time-major rows: n*B+b)
  k_embed<<<400, 256, 0, stream>>>(x, enc_emb, emb);
  // 2) input-gate precompute: (400,512)@(512,2048)+bias  [WMMA]
  k_gemm_bf16<0, false><<<dim3(32, 25), 128, 0, stream>>>(emb, wihFs, f_b, nullptr, xiF, 400, 2048, 2048, 512);
  k_gemm_bf16<0, false><<<dim3(32, 25), 128, 0, stream>>>(emb, wihBs, b_b, nullptr, xiB, 400, 2048, 2048, 512);
  // 3) zero initial h/c
  k_zero<<<32, 256, 0, stream>>>(h0, 16 * 512);
  k_zero<<<32, 256, 0, stream>>>(cF0, 16 * 512);
  k_zero<<<32, 256, 0, stream>>>(cB0, 16 * 512);
  // 4) LSTM forward (25 sequential WMMA steps)
  for (int t = 0; t < N_; ++t) {
    const float* hp = (t == 0) ? h0 : (hsF + (t - 1) * 16 * 512);
    float* cp = (t & 1) ? cF1 : cF0;
    float* cn = (t & 1) ? cF0 : cF1;
    k_lstm_step<<<8, 128, 0, stream>>>(xiF + (size_t)t * 16 * 2048, hp, cp, whhFs,
                                       hsF + t * 16 * 512, cn);
  }
  // 5) LSTM backward (consumes reversed time; stores un-reversed)
  for (int t = 0; t < N_; ++t) {
    int tm = N_ - 1 - t;
    const float* hp = (t == 0) ? h0 : (hsB + (tm + 1) * 16 * 512);
    float* cp = (t & 1) ? cB1 : cB0;
    float* cn = (t & 1) ? cB0 : cB1;
    k_lstm_step<<<8, 128, 0, stream>>>(xiB + (size_t)tm * 16 * 2048, hp, cp, whhBs,
                                       hsB + tm * 16 * 512, cn);
  }
  // 6) max-pool + encoder head + KL + reparameterized z (writes d_out[16..31])
  k_pool_enc<<<1, 128, 0, stream>>>(hsF, hsB, eps, encp_w, encp_b, z, (float*)d_out);
  // 7) build concatenated MLP inputs
  k_concat<<<960, 128, 0, stream>>>(t_emb, T_, z, tin, 960);
  k_concat<<<480, 128, 0, stream>>>(nt_emb, NT_, z, ntin, 480);
  k_concat<<<16, 128, 0, stream>>>(root_emb, 1, z, rin, 16);
  // 8) root MLP + log_softmax (tiny; single block)
  k_root_mlp<<<1, 256, 0, stream>>>(rin, root_w1, root_b1, root_rw, root_rb,
                                    root_w2, root_b2, roots);
  // 9) vocab trunk MLP (WMMA GEMM chain, residual structure)
  k_gemm_bf16<0, false><<<dim3(4, 60), 128, 0, stream>>>(tin, vw1s, voc_b1, nullptr, hA, 960, 256, 256, 320);
  k_gemm_bf16<1, false><<<dim3(4, 60), 128, 0, stream>>>(hA, vrws, voc_rb, nullptr, aA, 960, 256, 256, 256);
  k_gemm_bf16<1, true><<<dim3(4, 60), 128, 0, stream>>>(aA, vrws + 65536, voc_rb + 256, hA, hB2, 960, 256, 256, 256);
  k_gemm_bf16<1, false><<<dim3(4, 60), 128, 0, stream>>>(hB2, vrws + 2 * 65536, voc_rb + 512, nullptr, aA, 960, 256, 256, 256);
  k_gemm_bf16<1, true><<<dim3(4, 60), 128, 0, stream>>>(aA, vrws + 3 * 65536, voc_rb + 768, hB2, hA, 960, 256, 256, 256);
  // 10) fused (960,256)@(256,10000) + streaming logsumexp (no 38MB logits)
  k_vocab_lse<<<60, 128, 0, stream>>>(hA, vw2s, voc_b2, lse);
  // 11) unary gather: only the B*N needed vocab columns
  k_unary<<<400, 64, 0, stream>>>(hA, voc_w2, voc_b2, lse, x, un);
  // 12) rule scores: (480,320)@(320,8100) [WMMA] then row log_softmax
  k_gemm_bf16<0, false><<<dim3(127, 30), 128, 0, stream>>>(ntin, rulsw, rule_b, nullptr, rule, 480, 8100, 8112, 320);
  k_lsm_rows<<<480, 256, 0, stream>>>(rule, 8100);
  // 13) inside algorithm
  k_chart_init<<<(B_ * N_ * N_ * S_ + 255) / 256, 256, 0, stream>>>(chart, un);
  for (int w = 2; w <= N_; ++w)
    k_inside<<<dim3(N_ - w + 1, B_), 256, 0, stream>>>(chart, rule, w);
  // 14) -log_Z  (writes d_out[0..15])
  k_final<<<16, 32, 0, stream>>>(chart, roots, (float*)d_out);
}
